// MLP_Cutoff_87737591923195
// MI455X (gfx1250) — matrix operand order repaired and checked by reading, not compiled, not run on hardware
//
#include <hip/hip_runtime.h>

// ---------------------------------------------------------------------------
// 50-group batched tiny-MLP (DLRM-style cutoff net) for gfx1250 / MI455X.
//
// ~70 GFLOP vs ~6 MB unique bytes -> pure matrix-compute problem.
// Layers 2-4 run on V_WMMA_F32_16X16X32_F16 (8192 MAC/instr, fp32 accum).
// One wave owns a 16-row batch tile; a 128-thread WG owns 64 rows and loops
// over all 50 groups with cooperative LDS weight staging (f16, zero-padded).
// Fragment gathers use one base VGPR + constant offsets so DS loads use
// immediate offset fields (no per-fragment address VALU).
// ---------------------------------------------------------------------------

typedef _Float16 half_t;
typedef half_t v16h __attribute__((ext_vector_type(16)));
typedef float  v8f  __attribute__((ext_vector_type(8)));

// Padded sizes (N padded to mult of 16, K to mult of 32).
// Strides padded so the lane->bank map of fragment gathers is conflict-free.
constexpr int HSTR = 164;   // activation row stride in halfs (160 used + pad)
constexpr int WS2  = 164;   // W2 row stride (K=140 -> 160 used)
constexpr int WS3  = 132;   // W3 row stride (K=100 -> 128 used)
constexpr int WS4  = 68;    // W4 row stride (K=60  -> 64  used)

// Dynamic LDS layout (bytes)
constexpr int OFF_XS = 0;                        // [64][32] f32 x tile
constexpr int OFF_W1 = OFF_XS + 64 * 32 * 4;     // [140][2] f32
constexpr int OFF_B1 = OFF_W1 + 140 * 2 * 4;     // [144] f32
constexpr int OFF_B2 = OFF_B1 + 144 * 4;         // [112] f32
constexpr int OFF_B3 = OFF_B2 + 112 * 4;         // [64]  f32
constexpr int OFF_B4 = OFF_B3 + 64 * 4;          // [32]  f32
constexpr int OFF_W5 = OFF_B4 + 32 * 4;          // [20]  f32
constexpr int OFF_B5 = OFF_W5 + 20 * 4;          // [1]   f32 (+pad)
constexpr int OFF_W2 = OFF_B5 + 16;              // [112][WS2] f16
constexpr int OFF_W3 = OFF_W2 + 112 * WS2 * 2;   // [64][WS3]  f16
constexpr int OFF_W4 = OFF_W3 + 64 * WS3 * 2;    // [32][WS4]  f16
constexpr int OFF_HA = OFF_W4 + 32 * WS4 * 2;    // 4 waves * [16][HSTR] f16
constexpr int OFF_HB = OFF_HA + 4 * 16 * HSTR * 2;
constexpr int SMEM_BYTES = OFF_HB + 4 * 16 * HSTR * 2;   // = 110784 B

union AF { v16h v; unsigned u[8]; };

// One wave computes C[16 x NT*16] = relu( A[16 x KT*32] * W^T + bias ) in f16.
// Fragment element placement follows CDNA5 ISA 7.12.2:
//   A (16x32 f16):  m = lane&15;  k = 16*(r>>2) + 8*(lane>>4) + 2*(r&3) + j
//   B (32x16 f16):  n = lane&15;  k = 16*(r>>2) + 8*(lane>>4) + 2*(r&3) + j
//   C (16x16 f32):  m = r + 8*(lane>>4); n = lane&15
template <int NT, int KT, int WSTR>
__device__ __forceinline__ void gemm_relu(
    const half_t* __restrict__ hin,   // this wave's input acts [16][HSTR]
    const half_t* __restrict__ W,     // LDS weights [NT*16][WSTR], out-major
    const float*  __restrict__ bias,  // LDS [NT*16] (zero padded)
    half_t* __restrict__ hout,        // this wave's output acts [16][HSTR]
    int lane)
{
    const int nlo = lane & 15;
    const int hh  = lane >> 4;

    // Lane-dependent bases; every gather below is base + constant offset,
    // so DS loads can use their immediate offset field.
    const half_t* arow  = hin + nlo * HSTR + (hh << 3);
    const half_t* bbase = W   + nlo * WSTR + (hh << 3);

    v8f c[NT];
#pragma unroll
    for (int t = 0; t < NT; ++t) {
        float b = bias[t * 16 + nlo];          // C element depends on n only
        c[t] = (v8f){b, b, b, b, b, b, b, b};
    }

#pragma unroll
    for (int k0 = 0; k0 < KT; ++k0) {
        AF a;
#pragma unroll
        for (int hi = 0; hi < 2; ++hi) {       // r = hi*4 .. hi*4+3
            uint2 q0 = *(const uint2*)(arow + k0 * 32 + hi * 16);
            uint2 q1 = *(const uint2*)(arow + k0 * 32 + hi * 16 + 4);
            a.u[hi * 4 + 0] = q0.x; a.u[hi * 4 + 1] = q0.y;
            a.u[hi * 4 + 2] = q1.x; a.u[hi * 4 + 3] = q1.y;
        }
#pragma unroll
        for (int t = 0; t < NT; ++t) {
            AF bf;
#pragma unroll
            for (int r2 = 0; r2 < 4; ++r2) {   // two 8-half runs (k 8h.., 16+8h..) per (t,k0)
                uint2 q = *(const uint2*)(bbase + t * 16 * WSTR + k0 * 32 + ((r2 & 1) << 2) + ((r2 >> 1) << 4));
                bf.u[2 * r2 + 0] = q.x; bf.u[2 * r2 + 1] = q.y;
            }
            c[t] = __builtin_amdgcn_wmma_f32_16x16x32_f16(
                false, a.v, false, bf.v, (short)0, c[t], false, false);
        }
    }

    half_t* orow = hout + (hh << 3) * HSTR + nlo;
#pragma unroll
    for (int t = 0; t < NT; ++t) {
#pragma unroll
        for (int r = 0; r < 8; ++r) {
            float v = c[t][r];
            v = v > 0.0f ? v : 0.0f;
            orow[r * HSTR + t * 16] = (half_t)v;
        }
    }
}

__global__ void __launch_bounds__(128)
mlp_cutoff_kernel(const float* __restrict__ x, const int* __restrict__ inter_idx,
                  const float* mW1, const float* mb1, const float* iW1, const float* ib1,
                  const float* mW2, const float* mb2, const float* iW2, const float* ib2,
                  const float* mW3, const float* mb3, const float* iW3, const float* ib3,
                  const float* mW4, const float* mb4, const float* iW4, const float* ib4,
                  const float* mW5, const float* mb5, const float* iW5, const float* ib5,
                  float* __restrict__ out)
{
    extern __shared__ char smem[];
    float*  xs  = (float*)(smem + OFF_XS);
    float*  w1s = (float*)(smem + OFF_W1);
    float*  b1s = (float*)(smem + OFF_B1);
    float*  b2s = (float*)(smem + OFF_B2);
    float*  b3s = (float*)(smem + OFF_B3);
    float*  b4s = (float*)(smem + OFF_B4);
    float*  w5s = (float*)(smem + OFF_W5);
    float*  b5s = (float*)(smem + OFF_B5);
    half_t* w2s = (half_t*)(smem + OFF_W2);
    half_t* w3s = (half_t*)(smem + OFF_W3);
    half_t* w4s = (half_t*)(smem + OFF_W4);
    half_t* hA  = (half_t*)(smem + OFF_HA);
    half_t* hB  = (half_t*)(smem + OFF_HB);

    const int tid  = threadIdx.x;
    const int lane = tid & 31;
    const int wave = tid >> 5;
    const int rowbase = blockIdx.x * 64;
    const int m = lane & 15;

    // Stage this WG's 64x30 x-tile (fully coalesced global reads).
    for (int i = tid; i < 64 * 30; i += 128) {
        int r = i / 30, f = i % 30;
        xs[r * 32 + f] = x[(size_t)(rowbase + r) * 30 + f];
    }
    // Zero activation buffers once: keeps all K/N pad columns zero forever
    // (layers only ever rewrite the live columns).
    {
        unsigned* za = (unsigned*)hA;
        unsigned* zb = (unsigned*)hB;
        for (int i = tid; i < (4 * 16 * HSTR) / 2; i += 128) { za[i] = 0u; zb[i] = 0u; }
    }

    half_t* myA = hA + wave * 16 * HSTR;
    half_t* myB = hB + wave * 16 * HSTR;
    float yacc = 0.0f;

    for (int g = 0; g < 50; ++g) {
        const bool inter = (g >= 30);
        const int  gi    = inter ? g - 30 : g;
        const float* W1 = inter ? iW1 + gi * 140 * 2 : mW1 + gi * 140;
        const float* B1 = (inter ? ib1 : mb1) + gi * 140;
        const float* W2 = (inter ? iW2 : mW2) + gi * 100 * 140;
        const float* B2 = (inter ? ib2 : mb2) + gi * 100;
        const float* W3 = (inter ? iW3 : mW3) + gi * 60 * 100;
        const float* B3 = (inter ? ib3 : mb3) + gi * 60;
        const float* W4 = (inter ? iW4 : mW4) + gi * 20 * 60;
        const float* B4 = (inter ? ib4 : mb4) + gi * 20;
        const float* W5 = (inter ? iW5 : mW5) + gi * 20;
        const float* B5 = (inter ? ib5 : mb5) + gi;

        // ---- cooperative staging: fp32 global -> zero-padded f16 LDS ----
        for (int i = tid; i < 112 * WS2; i += 128) {
            int o = i / WS2, k = i % WS2;
            float v = (o < 100 && k < 140) ? W2[o * 140 + k] : 0.0f;
            w2s[i] = (half_t)v;
        }
        for (int i = tid; i < 64 * WS3; i += 128) {
            int o = i / WS3, k = i % WS3;
            float v = (o < 60 && k < 100) ? W3[o * 100 + k] : 0.0f;
            w3s[i] = (half_t)v;
        }
        for (int i = tid; i < 32 * WS4; i += 128) {
            int o = i / WS4, k = i % WS4;
            float v = (o < 20 && k < 60) ? W4[o * 60 + k] : 0.0f;
            w4s[i] = (half_t)v;
        }
        for (int i = tid; i < 140; i += 128) {
            if (inter) { w1s[2 * i] = W1[2 * i]; w1s[2 * i + 1] = W1[2 * i + 1]; }
            else       { w1s[2 * i] = W1[i];     w1s[2 * i + 1] = 0.0f; }
            b1s[i] = B1[i];
        }
        if (tid < 112) b2s[tid] = (tid < 100) ? B2[tid] : 0.0f;
        else if (tid >= 112 && tid < 112 + 64) {
            int t = tid - 112;
            b3s[t] = (t < 60) ? B3[t] : 0.0f;
        }
        if (tid < 32) b4s[tid] = (tid < 20) ? B4[tid] : 0.0f;
        if (tid >= 32 && tid < 52) w5s[tid - 32] = W5[tid - 32];
        if (tid == 63) b5s[0] = B5[0];
        __syncthreads();

        // ---- layer 1: in-dim 1 or 2, pure VALU outer product ----
        int f0, f1;
        if (inter) { f0 = inter_idx[2 * gi]; f1 = inter_idx[2 * gi + 1]; }
        else       { f0 = g; f1 = g; }      // w1 col1 is zero for main groups
        float x0 = xs[(wave * 16 + m) * 32 + f0];
        float x1 = xs[(wave * 16 + m) * 32 + f1];
        {
            const int ob = lane >> 4;       // two output neurons per iteration
            half_t* h1row = myA + m * HSTR;
            for (int i = 0; i < 70; ++i) {
                int o = ob + 2 * i;
                float h = fmaf(w1s[2 * o], x0, fmaf(w1s[2 * o + 1], x1, b1s[o]));
                h1row[o] = (half_t)(h > 0.0f ? h : 0.0f);
            }
        }

        // ---- layers 2-4: WMMA chains (fp32 accumulate, fused bias+relu) ----
        gemm_relu<7, 5, WS2>(myA, w2s, b2s, myB, lane);  // 140 -> 100
        gemm_relu<4, 4, WS3>(myB, w3s, b3s, myA, lane);  // 100 -> 60
        gemm_relu<2, 2, WS4>(myA, w4s, b4s, myB, lane);  //  60 -> 20

        // ---- layer 5: dot-20 + bias, accumulate per-row sum over groups ----
        float dot = b5s[0];
#pragma unroll
        for (int k = 0; k < 20; ++k) dot += (float)myB[m * HSTR + k] * w5s[k];
        yacc += dot;

        // Prefetch next group's dominant weight panel (global_prefetch_b8).
        if (g + 1 < 50) {
            const float* W2n = (g + 1 >= 30) ? iW2 + (size_t)(g + 1 - 30) * 100 * 140
                                             : mW2 + (size_t)(g + 1) * 100 * 140;
            for (int i = tid * 32; i < 14000; i += 128 * 32)
                __builtin_prefetch(W2n + i, 0, 3);
        }
        __syncthreads();   // weights may be overwritten next iteration
    }

    if (lane < 16) out[rowbase + wave * 16 + m] = yacc;
}

extern "C" void kernel_launch(void* const* d_in, const int* in_sizes, int n_in,
                              void* d_out, int out_size, void* d_ws, size_t ws_size,
                              hipStream_t stream)
{
    (void)n_in; (void)out_size; (void)d_ws; (void)ws_size;
    const float* x         = (const float*)d_in[0];
    const int*   inter_idx = (const int*)  d_in[1];
    const float* mW1 = (const float*)d_in[2];  const float* mb1 = (const float*)d_in[3];
    const float* iW1 = (const float*)d_in[4];  const float* ib1 = (const float*)d_in[5];
    const float* mW2 = (const float*)d_in[6];  const float* mb2 = (const float*)d_in[7];
    const float* iW2 = (const float*)d_in[8];  const float* ib2 = (const float*)d_in[9];
    const float* mW3 = (const float*)d_in[10]; const float* mb3 = (const float*)d_in[11];
    const float* iW3 = (const float*)d_in[12]; const float* ib3 = (const float*)d_in[13];
    const float* mW4 = (const float*)d_in[14]; const float* mb4 = (const float*)d_in[15];
    const float* iW4 = (const float*)d_in[16]; const float* ib4 = (const float*)d_in[17];
    const float* mW5 = (const float*)d_in[18]; const float* mb5 = (const float*)d_in[19];
    const float* iW5 = (const float*)d_in[20]; const float* ib5 = (const float*)d_in[21];
    float* out = (float*)d_out;

    const int batch = in_sizes[0] / 30;        // 32768
    const int grid  = batch / 64;              // 64 rows per 128-thread WG

    (void)hipFuncSetAttribute((const void*)mlp_cutoff_kernel,
                              hipFuncAttributeMaxDynamicSharedMemorySize,
                              (int)SMEM_BYTES);
    mlp_cutoff_kernel<<<dim3(grid), dim3(128), SMEM_BYTES, stream>>>(
        x, inter_idx,
        mW1, mb1, iW1, ib1, mW2, mb2, iW2, ib2,
        mW3, mb3, iW3, ib3, mW4, mb4, iW4, ib4,
        mW5, mb5, iW5, ib5, out);
}
